// GranularAgent_21311627723135
// MI455X (gfx1250) — matrix operand order repaired and checked by reading, not compiled, not run on hardware
//
#include <hip/hip_runtime.h>
#include <hip/hip_bf16.h>
#include <math.h>

// ---------------------------------------------------------------------------
// GranularAgent fused pipeline for gfx1250 (MI455X), wave32 + WMMA f16.
//   x0 = emb + 0.1*nb + mu
//   H  = silu(x0 @ Wup + bup)                (WMMA f16 -> f32 acc)
//   Y  = x0 + tanh(g)*(H @ Wdown + bdn)      (WMMA f16 -> f32 acc)
//   j  = argmin_j ||Y - C_j||  via  cnorm_j - 2*Y.C_j^T   (WMMA)
//   out = norm_clamp(nan_to_num(Y + 0.05*C_j), tau=10)
// All intermediates for a 32-row tile live in LDS (~162 KB / WGP).
// Each wave owns a full 32-row column stripe: B fragments are loaded once and
// used by two WMMAs (rt=0/rt=1), halving L2 weight traffic.
// ---------------------------------------------------------------------------

typedef __attribute__((ext_vector_type(16))) _Float16 v16h;
typedef __attribute__((ext_vector_type(8)))  _Float16 v8h;
typedef __attribute__((ext_vector_type(4)))  _Float16 v4h;
typedef __attribute__((ext_vector_type(8)))  float    v8f;

#define NROWS   131072
#define DIM     512
#define DIM2    1024
#define MEMN    512

#define M_TILE  32
#define A_STR   520    // f16 elems, padded (1040 B rows -> conflict-free b128)
#define R_STR   516    // f32 elems, padded
#define H_STR   1032   // f16 elems, padded

#define LDS_A_OFF   0u
#define LDS_R_OFF   33280u                       // 32*520*2
#define LDS_H_OFF   (33280u + 66048u)            // + 32*516*4
#define LDS_KEY_OFF (33280u + 66048u + 66048u)   // + 32*1032*2  = 165376
#define LDS_TOTAL   (LDS_KEY_OFF + 32u * 8u)     // + 32 u64 keys = 165632

// workspace layout (bytes)
#define WS_WUPT  0u           // f16 [1024][512]  (Wup transposed, N-major)
#define WS_WDNT  1048576u     // f16 [512][1024]  (Wdown transposed, N-major)
#define WS_CF16  2097152u     // f16 [512][512]   (memory_book, already N-major)
#define WS_CNORM 2621440u     // f32 [512]        (||c||^2)

// ----- WMMA fragment loaders (contiguous k = 16h + s order, same for A and B, wave32) -------------------
// A (16x32 f16): lane L holds row M=L&15; K = (L>=16?16:0) + 0..15
__device__ __forceinline__ v16h load_a_frag(const _Float16* base, int stride,
                                            int row0, int kb, int lane) {
  const _Float16* p = base + (row0 + (lane & 15)) * stride
                           + kb + ((lane >> 4) << 4);
  v8h lo = *(const v8h*)(p);        // K = kb + h16 + 0..7
  v8h hi = *(const v8h*)(p + 8);    // K = kb + h16 + 8..15
  return __builtin_shufflevector(lo, hi, 0,1,2,3,4,5,6,7,8,9,10,11,12,13,14,15);
}

// B (32x16 f16): lane L holds col N=L&15; K = kb + (L>=16?16:0) + 0..15.
// Weights are stored transposed [N][K] so this is 32 contiguous bytes.
__device__ __forceinline__ v16h load_b_frag(const _Float16* base, int kstride,
                                            int n0, int kb, int lane) {
  const _Float16* p = base + (size_t)(n0 + (lane & 15)) * kstride
                           + kb + ((lane >> 4) << 4);
  return *(const v16h*)p;
}

// ----- prep: f32 -> f16 weight transposes + codebook norms ------------------
extern "C" __global__ __launch_bounds__(256)
void ga_prep(const float* __restrict__ wup, const float* __restrict__ wdn,
             const float* __restrict__ cbook,
             _Float16* __restrict__ wupt, _Float16* __restrict__ wdnt,
             _Float16* __restrict__ cf16) {
  unsigned i = blockIdx.x * 256u + threadIdx.x;
  if (i < 524288u) {                       // Wup [512][1024] -> [1024][512]
    unsigned n = i >> 9, k = i & 511u;
    wupt[i] = (_Float16)wup[k * 1024u + n];
  } else if (i < 1048576u) {               // Wdown [1024][512] -> [512][1024]
    unsigned j = i - 524288u;
    unsigned n = j >> 10, k = j & 1023u;
    wdnt[j] = (_Float16)wdn[k * 512u + n];
  } else if (i < 1310720u) {               // memory_book copy-convert
    unsigned j = i - 1048576u;
    cf16[j] = (_Float16)cbook[j];
  }
}

extern "C" __global__ __launch_bounds__(256)
void ga_cnorm(const float* __restrict__ cbook, float* __restrict__ cnorm) {
  __shared__ float red[256];
  const int b = blockIdx.x, t = threadIdx.x;
  float v0 = cbook[b * 512 + t];
  float v1 = cbook[b * 512 + 256 + t];
  red[t] = v0 * v0 + v1 * v1;
  __syncthreads();
  for (int s = 128; s > 0; s >>= 1) {
    if (t < s) red[t] += red[t + s];
    __syncthreads();
  }
  if (t == 0) cnorm[b] = red[0];
}

// ----- fused main kernel: one 32-row tile per block (256 thr = 8 waves) -----
extern "C" __global__ __launch_bounds__(256)
void ga_fused(const float* __restrict__ emb, const float* __restrict__ nb,
              const float* __restrict__ mu,  const float* __restrict__ bup,
              const float* __restrict__ bdn, const float* __restrict__ gate,
              const float* __restrict__ cbook,
              const _Float16* __restrict__ wupt,
              const _Float16* __restrict__ wdnt,
              const _Float16* __restrict__ cf16,
              const float* __restrict__ cnorm,
              float* __restrict__ out) {
  extern __shared__ char smem[];
  _Float16* ldsA = (_Float16*)(smem + LDS_A_OFF);   // x0 / Y as f16 [32][520]
  float*    ldsR = (float*)   (smem + LDS_R_OFF);   // x0 / Y as f32 [32][516]
  _Float16* ldsH = (_Float16*)(smem + LDS_H_OFF);   // H f16        [32][1032]
  unsigned long long* ldsKey = (unsigned long long*)(smem + LDS_KEY_OFF);

  const int tid  = threadIdx.x;
  const int lane = tid & 31;
  const int wave = tid >> 5;
  const int row_base = blockIdx.x * M_TILE;

  // ---- Phase A: x0 = emb + 0.1*nb + mu -> LDS (f32 + f16) ----
  const float4* emb4 = (const float4*)emb;
  const float4* nb4  = (const float4*)nb;
  const float4* mu4  = (const float4*)mu;
  for (int u = tid; u < M_TILE * (DIM / 4); u += 256) {
    int r = u >> 7, c4 = u & 127;
    size_t gi = (size_t)(row_base + r) * (DIM / 4) + c4;
    float4 e = emb4[gi], b = nb4[gi], m = mu4[c4];
    float4 x;
    x.x = e.x + 0.1f * b.x + m.x;
    x.y = e.y + 0.1f * b.y + m.y;
    x.z = e.z + 0.1f * b.z + m.z;
    x.w = e.w + 0.1f * b.w + m.w;
    int col = c4 << 2;
    *(float4*)&ldsR[r * R_STR + col] = x;
    v4h hx = { (_Float16)x.x, (_Float16)x.y, (_Float16)x.z, (_Float16)x.w };
    *(v4h*)&ldsA[r * A_STR + col] = hx;
  }
  __syncthreads();

  // ---- Phase B: H = silu(x0 @ Wup + bup) ----
  // 64 column stripes of 32 rows; 8 stripes/wave; B loaded once per k-step.
  for (int ct = wave; ct < 64; ct += 8) {
    v8f acc0 = {}, acc1 = {};
#pragma unroll 4
    for (int kb = 0; kb < DIM; kb += 32) {
      v16h b  = load_b_frag(wupt, DIM, ct * 16, kb, lane);
      v16h a0 = load_a_frag(ldsA, A_STR, 0,  kb, lane);
      v16h a1 = load_a_frag(ldsA, A_STR, 16, kb, lane);
      acc0 = __builtin_amdgcn_wmma_f32_16x16x32_f16(false, a0, false, b,
                                                    (short)0, acc0, false, false);
      acc1 = __builtin_amdgcn_wmma_f32_16x16x32_f16(false, a1, false, b,
                                                    (short)0, acc1, false, false);
    }
    const int n = ct * 16 + (lane & 15);
    const float bb = bup[n];
    const int hi8 = (lane >> 4) << 3;
#pragma unroll
    for (int v = 0; v < 8; v++) {
      float h0 = acc0[v] + bb;
      float h1 = acc1[v] + bb;
      // SiLU with fast hardware reciprocal (v_rcp_f32, no div fixup chain)
      float s0 = h0 * __builtin_amdgcn_rcpf(1.0f + __expf(-h0));
      float s1 = h1 * __builtin_amdgcn_rcpf(1.0f + __expf(-h1));
      ldsH[(hi8 + v)      * H_STR + n] = (_Float16)s0;
      ldsH[(16 + hi8 + v) * H_STR + n] = (_Float16)s1;
    }
  }
  __syncthreads();

  // ---- Phase C: Y = x0 + tanh(g)*(H @ Wdown + bdn) ---- (32 stripes, 4/wave)
  const float tg = tanhf(gate[0]);
  for (int ct = wave; ct < 32; ct += 8) {
    v8f acc0 = {}, acc1 = {};
#pragma unroll 4
    for (int kb = 0; kb < DIM2; kb += 32) {
      v16h b  = load_b_frag(wdnt, DIM2, ct * 16, kb, lane);
      v16h a0 = load_a_frag(ldsH, H_STR, 0,  kb, lane);
      v16h a1 = load_a_frag(ldsH, H_STR, 16, kb, lane);
      acc0 = __builtin_amdgcn_wmma_f32_16x16x32_f16(false, a0, false, b,
                                                    (short)0, acc0, false, false);
      acc1 = __builtin_amdgcn_wmma_f32_16x16x32_f16(false, a1, false, b,
                                                    (short)0, acc1, false, false);
    }
    const int n = ct * 16 + (lane & 15);
    const float bb = bdn[n];
    const int hi8 = (lane >> 4) << 3;
#pragma unroll
    for (int v = 0; v < 8; v++) {
      const int m0 = hi8 + v, m1 = 16 + hi8 + v;
      float y0 = ldsR[m0 * R_STR + n] + tg * (acc0[v] + bb);
      float y1 = ldsR[m1 * R_STR + n] + tg * (acc1[v] + bb);
      ldsR[m0 * R_STR + n] = y0;            // Y f32 (stripe-owned slot)
      ldsR[m1 * R_STR + n] = y1;
      ldsA[m0 * A_STR + n] = (_Float16)y0;  // Y f16 for score GEMM
      ldsA[m1 * A_STR + n] = (_Float16)y1;
    }
  }
  if (tid < 32) ldsKey[tid] = ~0ull;
  __syncthreads();

  // ---- Phase D: scores = cnorm - 2*Y.C^T ; per-row argmin via ds_min_u64 ----
  for (int ct = wave; ct < 32; ct += 8) {
    v8f acc0 = {}, acc1 = {};
#pragma unroll 4
    for (int kb = 0; kb < DIM; kb += 32) {
      v16h b  = load_b_frag(cf16, DIM, ct * 16, kb, lane);
      v16h a0 = load_a_frag(ldsA, A_STR, 0,  kb, lane);
      v16h a1 = load_a_frag(ldsA, A_STR, 16, kb, lane);
      acc0 = __builtin_amdgcn_wmma_f32_16x16x32_f16(false, a0, false, b,
                                                    (short)0, acc0, false, false);
      acc1 = __builtin_amdgcn_wmma_f32_16x16x32_f16(false, a1, false, b,
                                                    (short)0, acc1, false, false);
    }
    const int j = ct * 16 + (lane & 15);
    const float cn = cnorm[j];
    const int hi8 = (lane >> 4) << 3;
#pragma unroll
    for (int v = 0; v < 8; v++) {
      float s0 = cn - 2.0f * acc0[v];
      float s1 = cn - 2.0f * acc1[v];
      unsigned int ub0  = __float_as_uint(s0);
      unsigned int ub1  = __float_as_uint(s1);
      unsigned int ord0 = (ub0 & 0x80000000u) ? ~ub0 : (ub0 | 0x80000000u);
      unsigned int ord1 = (ub1 & 0x80000000u) ? ~ub1 : (ub1 | 0x80000000u);
      unsigned long long k0 = ((unsigned long long)ord0 << 32) | (unsigned int)j;
      unsigned long long k1 = ((unsigned long long)ord1 << 32) | (unsigned int)j;
      atomicMin(&ldsKey[hi8 + v], k0);      // min score, then min index
      atomicMin(&ldsKey[16 + hi8 + v], k1);
    }
  }
  __syncthreads();

  // ---- Phase E: gather codeword, nan_to_num, norm clamp, store ----
  for (int rr = 0; rr < 4; rr++) {
    const int r = wave * 4 + rr;
    const unsigned long long key = ldsKey[r];
    const int idx = (int)(unsigned int)(key & 0xFFFFFFFFull);
    const float* crow = cbook + (size_t)idx * DIM;
    float xs[16];
    float ss = 0.0f;
#pragma unroll
    for (int i = 0; i < 16; i++) {
      const int col = i * 32 + lane;
      float x = ldsR[r * R_STR + col] + 0.05f * crow[col];
      if (x != x) x = 0.0f;                           // nan -> 0
      else if (isinf(x)) x = (x > 0.0f) ? 1.0f : -1.0f;
      xs[i] = x;
      ss += x * x;
    }
#pragma unroll
    for (int off = 16; off > 0; off >>= 1) ss += __shfl_xor(ss, off, 32);
    const float nrm = sqrtf(ss);
    const float sc = (nrm > 10.0f) ? (10.0f / fmaxf(nrm, 1e-6f)) : 1.0f;
    const size_t ob = (size_t)(row_base + r) * DIM;
#pragma unroll
    for (int i = 0; i < 16; i++) out[ob + i * 32 + lane] = xs[i] * sc;
  }
}

// ---------------------------------------------------------------------------
extern "C" void kernel_launch(void* const* d_in, const int* in_sizes, int n_in,
                              void* d_out, int out_size, void* d_ws, size_t ws_size,
                              hipStream_t stream) {
  const float* emb   = (const float*)d_in[0];
  const float* nb    = (const float*)d_in[1];
  const float* mu    = (const float*)d_in[2];
  const float* wup   = (const float*)d_in[3];
  const float* bup   = (const float*)d_in[4];
  const float* wdn   = (const float*)d_in[5];
  const float* bdn   = (const float*)d_in[6];
  const float* gate  = (const float*)d_in[7];
  const float* cbook = (const float*)d_in[8];
  float* out = (float*)d_out;

  char* ws = (char*)d_ws;
  _Float16* wupt  = (_Float16*)(ws + WS_WUPT);
  _Float16* wdnt  = (_Float16*)(ws + WS_WDNT);
  _Float16* cf16  = (_Float16*)(ws + WS_CF16);
  float*    cnorm = (float*)(ws + WS_CNORM);

  ga_prep<<<5120, 256, 0, stream>>>(wup, wdn, cbook, wupt, wdnt, cf16);
  ga_cnorm<<<512, 256, 0, stream>>>(cbook, cnorm);
  ga_fused<<<NROWS / M_TILE, 256, LDS_TOTAL, stream>>>(
      emb, nb, mu, bup, bdn, gate, cbook, wupt, wdnt, cf16, cnorm, out);
}
